// scaled_dot_product_attention_56564719288366
// MI455X (gfx1250) — matrix operand order repaired and checked by reading, not compiled, not run on hardware
//
#include <hip/hip_runtime.h>
#include <hip/hip_bf16.h>

// Problem constants (from reference): B=16, Q=2048, K=2048, D=128
#define BATCH 16
#define QLEN  2048
#define KLEN  2048
#define DHEAD 128

#define KPITCH 136   // padded f16 pitch for key-major K tile  (272B row -> 68-bank stride)
#define VPITCH 72    // padded f16 pitch for d-major   V tile  (144B row -> 36-bank stride)

typedef _Float16 v16h __attribute__((ext_vector_type(16)));
typedef _Float16 v8h  __attribute__((ext_vector_type(8)));
typedef float    v8f  __attribute__((ext_vector_type(8)));
typedef float    v4f  __attribute__((ext_vector_type(4)));

static __device__ __forceinline__ v16h cat16(v8h lo, v8h hi) {
    v16h r;
#pragma unroll
    for (int i = 0; i < 8; ++i) { r[i] = lo[i]; r[i + 8] = hi[i]; }
    return r;
}

// Raw v_exp_f32 (base-2). No denormal fixup: inputs are <= 0 and huge negative
// masked scores are *supposed* to flush to zero, so the raw instruction is exact
// for softmax purposes and avoids the v_ldexp_f32 fixup sequence of exp2f().
static __device__ __forceinline__ float rexp2(float x) {
    return __builtin_amdgcn_exp2f(x);
}

// Flash attention, one 128-row Q tile per workgroup (8 waves x 16 rows),
// KV streamed in 64-key blocks through padded LDS (f16), WMMA f32_16x16x32_f16.
// Scores computed as S^T = K*Q^T so the softmax axis = M; key-row permutation
// makes the exp'ed C registers directly form the PV B-operand (no shuffles).
__global__ __launch_bounds__(256, 4)
void fa_fwd_wmma64_kernel(const float* __restrict__ Qp,
                          const float* __restrict__ Kp,
                          const float* __restrict__ Vp,
                          const int*   __restrict__ VL,
                          float* __restrict__ Op)
{
    __shared__ __align__(16) _Float16 ldsK[64 * KPITCH];     // key-major, padded
    __shared__ __align__(16) _Float16 ldsV[DHEAD * VPITCH];  // d-major (transposed), padded

    const int b    = blockIdx.y;
    const int qt   = blockIdx.x;
    const int tid  = threadIdx.x;
    const int wave = tid >> 5;
    const int lane = tid & 31;
    const int row  = lane & 15;
    const int half = lane >> 4;

    const int vl = VL[b];

    const int    q    = qt * 128 + wave * 16 + row;
    const float* qrow = Qp + ((size_t)b * QLEN + q) * DHEAD;

    // scale = log2(e)/sqrt(128), folded into Q before f16 conversion:
    // WMMA scores come out in base-2 units -> bare v_exp_f32.
    const float sc2 = 0.1275174217f;

    // Q fragments: B-operand of S^T (lane = query col, elem j <-> d = 32f + 16h + j).
    v16h qf[4];
#pragma unroll
    for (int f = 0; f < 4; ++f) {
        const int d0 = 32 * f + 16 * half;
        v4f x0 = *(const v4f*)(qrow + d0);
        v4f x1 = *(const v4f*)(qrow + d0 + 4);
        v4f x2 = *(const v4f*)(qrow + d0 + 8);
        v4f x3 = *(const v4f*)(qrow + d0 + 12);
        v16h t;
#pragma unroll
        for (int i = 0; i < 4; ++i) {
            t[i]      = (_Float16)(x0[i] * sc2);
            t[i + 4]  = (_Float16)(x1[i] * sc2);
            t[i + 8]  = (_Float16)(x2[i] * sc2);
            t[i + 12] = (_Float16)(x3[i] * sc2);
        }
        qf[f] = t;
    }

    // O^T accumulators: 8 tiles of 16(d) x 16(q); C layout: d = 16t + r + 8h.
    v8f of[8] = {};
    float m_run = -3.0e38f;
    float l_run = 0.0f;

    const float* kbase = Kp + (size_t)b * KLEN * DHEAD;
    const float* vbase = Vp + (size_t)b * KLEN * DHEAD;

    // Key-row permutation per 32-key group: tile0 rows = {0..7,16..23}, tile1 = {8..15,24..31}
    const int kr0 = row + (row & 8);
    const int kr1 = kr0 + 8;

    const int nblk = (vl + 63) >> 6;
    for (int i = 0; i < nblk; ++i) {
        const int kb = i * 64;

        // ---- cooperative stage: 64 keys x 128 d of K and V into padded LDS ----
        // K: key-major rows, pkrtz-packed b64 stores.
#pragma unroll
        for (int j = 0; j < 8; ++j) {
            const int e   = (tid + 256 * j) * 4;
            const int key = e >> 7;
            const int d   = e & 127;
            v4f kv = *(const v4f*)(kbase + (size_t)kb * DHEAD + e);
            auto lo = __builtin_amdgcn_cvt_pkrtz(kv[0], kv[1]);
            auto hi = __builtin_amdgcn_cvt_pkrtz(kv[2], kv[3]);
            auto k4 = __builtin_shufflevector(lo, hi, 0, 1, 2, 3);
            *(decltype(k4)*)&ldsK[key * KPITCH + d] = k4;
        }
        // V: transposed (d-major); each unit = one key-pair x 4 d -> 4 packed b32 stores.
#pragma unroll
        for (int j = 0; j < 4; ++j) {
            const int c  = tid + 256 * j;   // 1024 units = 32 key-pairs x 32 d-chunks
            const int kp = c >> 5;          // key pair index (keys 2kp, 2kp+1)
            const int d0 = (c & 31) * 4;
            v4f r0 = *(const v4f*)(vbase + (size_t)(kb + 2 * kp) * DHEAD + d0);
            v4f r1 = *(const v4f*)(vbase + (size_t)(kb + 2 * kp + 1) * DHEAD + d0);
#pragma unroll
            for (int k = 0; k < 4; ++k) {
                auto pk = __builtin_amdgcn_cvt_pkrtz(r0[k], r1[k]);
                *(decltype(pk)*)&ldsV[(d0 + k) * VPITCH + 2 * kp] = pk;
            }
        }
        // Prefetch next KV block into L2 while we compute (global_prefetch_b8).
        if (((tid & 7) == 0) && (kb + 64 < KLEN)) {
#pragma unroll
            for (int j = 0; j < 8; ++j) {
                const int e = (tid + 256 * j) * 4;
                __builtin_prefetch(kbase + (size_t)(kb + 64) * DHEAD + e, 0, 1);
                __builtin_prefetch(vbase + (size_t)(kb + 64) * DHEAD + e, 0, 1);
            }
        }
        __syncthreads();

        // ---- S^T = K_tile * Q^T : four 16x16 f32 C tiles (2 groups of 32 keys) ----
        v8f s[4] = {};
#pragma unroll
        for (int g = 0; g < 2; ++g) {
#pragma unroll
            for (int f = 0; f < 4; ++f) {
                const int off = 32 * f + 16 * half;
                v8h a0lo = *(const v8h*)&ldsK[(kr0 + 32 * g) * KPITCH + off];
                v8h a0hi = *(const v8h*)&ldsK[(kr0 + 32 * g) * KPITCH + off + 8];
                s[2 * g] = __builtin_amdgcn_wmma_f32_16x16x32_f16(
                    false, cat16(a0lo, a0hi), false, qf[f], (short)0, s[2 * g], false, false);
                v8h a1lo = *(const v8h*)&ldsK[(kr1 + 32 * g) * KPITCH + off];
                v8h a1hi = *(const v8h*)&ldsK[(kr1 + 32 * g) * KPITCH + off + 8];
                s[2 * g + 1] = __builtin_amdgcn_wmma_f32_16x16x32_f16(
                    false, cat16(a1lo, a1hi), false, qf[f], (short)0, s[2 * g + 1], false, false);
            }
        }

        // ---- length mask (only in the final, workgroup-uniform partial block) ----
        // Permuted mapping: s[2g] reg r <-> key kb + 32g + 16h + r; s[2g+1] <-> +8.
        if (kb + 64 > vl) {
            const int c0 = vl - kb - 16 * half;  // s[0] reg r valid iff r < c0
#pragma unroll
            for (int r = 0; r < 8; ++r) {
                if (r      >= c0) s[0][r] = -1.4426950e6f;
                if (r + 8  >= c0) s[1][r] = -1.4426950e6f;
                if (r + 32 >= c0) s[2][r] = -1.4426950e6f;
                if (r + 40 >= c0) s[3][r] = -1.4426950e6f;
            }
        }

        // ---- online softmax over 64 keys (per query = lane%16) ----
        float bmax = -3.0e38f;
#pragma unroll
        for (int g = 0; g < 4; ++g)
#pragma unroll
            for (int r = 0; r < 8; ++r) bmax = fmaxf(bmax, s[g][r]);
        bmax = fmaxf(bmax, __shfl_xor(bmax, 16, 32));
        const float mn   = fmaxf(m_run, bmax);
        const float corr = rexp2(m_run - mn);

        float lsum = 0.0f;
        v16h p[2];  // PV B-operand per group: elem j <-> key kb + 32g + 16h + j
#pragma unroll
        for (int g = 0; g < 2; ++g) {
#pragma unroll
            for (int r = 0; r < 8; ++r) {
                const float e0 = rexp2(s[2 * g][r] - mn);
                const float e1 = rexp2(s[2 * g + 1][r] - mn);
                lsum += e0 + e1;
                p[g][r]     = (_Float16)e0;
                p[g][r + 8] = (_Float16)e1;
            }
        }
        lsum += __shfl_xor(lsum, 16, 32);
        l_run = l_run * corr + lsum;
        m_run = mn;

        // ---- O^T += V^T * P^T : 8 d-tiles x 2 key groups ----
#pragma unroll
        for (int t = 0; t < 8; ++t) {
#pragma unroll
            for (int r = 0; r < 8; ++r) of[t][r] *= corr;
            const int vrow = (16 * t + row) * VPITCH;
#pragma unroll
            for (int g = 0; g < 2; ++g) {
                v8h vlo = *(const v8h*)&ldsV[vrow + 32 * g + 16 * half];
                v8h vhi = *(const v8h*)&ldsV[vrow + 32 * g + 16 * half + 8];
                of[t] = __builtin_amdgcn_wmma_f32_16x16x32_f16(
                    false, cat16(vlo, vhi), false, p[g], (short)0, of[t], false, false);
            }
        }
        __syncthreads();
    }

    // ---- epilogue: normalize and store O (lane's query row, d = 16t + r + 8h) ----
    const float invl = 1.0f / l_run;
    float* orow = Op + ((size_t)b * QLEN + q) * DHEAD;
#pragma unroll
    for (int t = 0; t < 8; ++t) {
        v4f lo, hi;
#pragma unroll
        for (int i = 0; i < 4; ++i) {
            lo[i] = of[t][i] * invl;
            hi[i] = of[t][i + 4] * invl;
        }
        *(v4f*)(orow + 16 * t + 8 * half)     = lo;
        *(v4f*)(orow + 16 * t + 8 * half + 4) = hi;
    }
}

extern "C" void kernel_launch(void* const* d_in, const int* in_sizes, int n_in,
                              void* d_out, int out_size, void* d_ws, size_t ws_size,
                              hipStream_t stream) {
    const float* Qp = (const float*)d_in[0];
    const float* Kp = (const float*)d_in[1];
    const float* Vp = (const float*)d_in[2];
    const int*   VL = (const int*)d_in[3];
    float*       Op = (float*)d_out;

    dim3 grid(QLEN / 128, BATCH);
    dim3 block(256);
    fa_fwd_wmma64_kernel<<<grid, block, 0, stream>>>(Qp, Kp, Vp, VL, Op);
}
